// HierarchicalRNN_10273561772586
// MI455X (gfx1250) — compile-verified, matrix operand order checked by reading, not run on hardware
//
#include <hip/hip_runtime.h>
#include <hip/hip_bf16.h>
#include <math.h>
#include <stdint.h>
#include <stddef.h>

typedef __bf16 bf16_t;
typedef __attribute__((ext_vector_type(8)))  __bf16 v8bf;
typedef __attribute__((ext_vector_type(16))) __bf16 v16bf;
typedef __attribute__((ext_vector_type(8)))  float   v8f;
typedef __attribute__((ext_vector_type(4)))  unsigned int u32x4;
typedef __attribute__((ext_vector_type(4)))  int          i32x4;
typedef __attribute__((ext_vector_type(8)))  int          i32x8;

// ---------------- numeric helpers ----------------
__device__ __forceinline__ bf16_t f2bf(float f) {
  union { float f; unsigned u; } v; v.f = f;
  unsigned r = v.u + 0x7fffu + ((v.u >> 16) & 1u);   // round-to-nearest-even
  union { unsigned short s; bf16_t b; } o; o.s = (unsigned short)(r >> 16);
  return o.b;
}
__device__ __forceinline__ float bf2f(bf16_t b) {
  union { unsigned short s; bf16_t b; } i; i.b = b;
  union { float f; unsigned u; } o; o.u = ((unsigned)i.s) << 16;
  return o.f;
}
__device__ __forceinline__ float sigmoidf_(float x) { return 1.0f / (1.0f + expf(-x)); }

// ------------- WMMA fragment helpers (16x16x32 bf16, wave32) -------------
// A[M,K] row-major and BT[N,K] row-major use the SAME per-lane K-vector layout
// (ISA 7.12.2): lanes 0-15 hold K[k0..k0+7] and K[k0+16..k0+23];
// lanes 16-31 hold K[k0+8..k0+15] and K[k0+24..k0+31].
__device__ __forceinline__ v16bf load_frag(const bf16_t* __restrict__ base, int ld,
                                           int row0, int k0, int lane) {
  const int kh = lane >> 4;
  const int r  = lane & 15;
  const bf16_t* p = base + (size_t)(row0 + r) * (size_t)ld + (size_t)(k0 + kh * 8);
  v8bf lo = *(const v8bf*)(p);
  v8bf hi = *(const v8bf*)(p + 16);
  return __builtin_shufflevector(lo, hi, 0,1,2,3,4,5,6,7,8,9,10,11,12,13,14,15);
}
__device__ __forceinline__ v8f wmma_bf16(v16bf a, v16bf b, v8f c) {
  return __builtin_amdgcn_wmma_f32_16x16x32_bf16(false, a, false, b, (short)0, c, false, false);
}

// ------------- generic WMMA GEMM with TDM weight staging -------------
// C[M,N] = act( A[M,K] * BT[N,K]^T + bias[N] ), M%64==0, N%128==0, K%32==0,
// ldb must equal K (weight panels are dense). The 128xK BT panel for this
// block is DMA'd into LDS by the Tensor Data Mover (TENSORcnt), then WMMA
// B-fragments are read from LDS while A streams from global.
// Block = 8 waves; each wave owns a 32x32 macro-tile (2x2 WMMA tiles).
__global__ void __launch_bounds__(256)
gemm_bf16_wmma(const bf16_t* __restrict__ A, int lda,
               const bf16_t* __restrict__ BT, int ldb,
               const float*  __restrict__ bias,
               float* __restrict__ Cf, bf16_t* __restrict__ Cb,
               int N, int K, int act)
{
  extern __shared__ bf16_t ldsB[];                 // [128][K] bf16 panel
  const int lane = threadIdx.x & 31;
  const int wave = threadIdx.x >> 5;
  const int row0 = blockIdx.y * 64  + (wave >> 2) * 32;
  const int col_base = blockIdx.x * 128;
  const int lcol0 = (wave & 3) * 32;               // local column of this wave's tile
  const int col0 = col_base + lcol0;
  const int kh = lane >> 4;
  const int lc = lane & 15;

  // ---- TDM: stage BT rows [col_base, col_base+128) x K into LDS ----
  if (wave == 0) {
    unsigned long long ga =
        (unsigned long long)(const void*)(BT + (size_t)col_base * (size_t)ldb);
    const unsigned kd = (unsigned)K;               // elements (bf16)
    u32x4 g0;
    g0[0] = 1u;                                    // count=1, user-mode
    g0[1] = 0u;                                    // lds_addr = 0 (panel base)
    g0[2] = (unsigned)(ga & 0xffffffffu);          // global_addr[31:0]
    g0[3] = (unsigned)((ga >> 32) & 0x01ffffffu)   // global_addr[56:32]
          | (2u << 30);                            // type = 2 (image/tensor)
    i32x8 g1;
    g1[0] = (int)(1u << 16);                       // wg_mask=0, data_size=1 (2B)
    g1[1] = (int)((kd & 0xffffu) << 16);           // tensor_dim0[15:0]
    g1[2] = (int)(((kd >> 16) & 0xffffu))          // tensor_dim0[31:16]
          | (int)(128u << 16);                     // tensor_dim1[15:0] = 128 rows
    g1[3] = (int)((kd & 0xffffu) << 16);           // tensor_dim1[31:16]=0, tile_dim0=K
    g1[4] = (int)128;                              // tile_dim1 = 128, tile_dim2 = 0
    g1[5] = (int)kd;                               // tensor_dim0_stride[31:0] = K
    g1[6] = 0;                                     // stride0[47:32]=0, stride1 lo=0
    g1[7] = 0;                                     // stride1 hi = 0
    i32x4 g2 = {0, 0, 0, 0};
    i32x4 g3 = {0, 0, 0, 0};
#if __clang_major__ >= 23
    i32x8 g4 = {0, 0, 0, 0, 0, 0, 0, 0};
    __builtin_amdgcn_tensor_load_to_lds(g0, g1, g2, g3, g4, 0);
#else
    __builtin_amdgcn_tensor_load_to_lds(g0, g1, g2, g3, 0);
#endif
    __builtin_amdgcn_s_wait_tensorcnt(0);
  }
  __syncthreads();

  const float b0 = bias ? bias[col0 + lc]      : 0.0f;
  const float b1 = bias ? bias[col0 + 16 + lc] : 0.0f;
  v8f c00, c01, c10, c11;
  #pragma unroll
  for (int i = 0; i < 8; ++i) { c00[i] = b0; c01[i] = b1; c10[i] = b0; c11[i] = b1; }

  for (int k0 = 0; k0 < K; k0 += 32) {
    v16bf a0 = load_frag(A, lda, row0,      k0, lane);
    v16bf a1 = load_frag(A, lda, row0 + 16, k0, lane);
    __builtin_prefetch(A + (size_t)(row0 + lc) * (size_t)lda + (size_t)(k0 + 32));
    // B fragments from the LDS panel (row stride = K elements)
    const bf16_t* pb0 = ldsB + (size_t)(lcol0 + lc)      * (size_t)K + (size_t)(k0 + kh * 8);
    const bf16_t* pb1 = ldsB + (size_t)(lcol0 + 16 + lc) * (size_t)K + (size_t)(k0 + kh * 8);
    v8bf w0lo = *(const v8bf*)(pb0);
    v8bf w0hi = *(const v8bf*)(pb0 + 16);
    v8bf w1lo = *(const v8bf*)(pb1);
    v8bf w1hi = *(const v8bf*)(pb1 + 16);
    v16bf w0 = __builtin_shufflevector(w0lo, w0hi, 0,1,2,3,4,5,6,7,8,9,10,11,12,13,14,15);
    v16bf w1 = __builtin_shufflevector(w1lo, w1hi, 0,1,2,3,4,5,6,7,8,9,10,11,12,13,14,15);
    c00 = wmma_bf16(a0, w0, c00);
    c01 = wmma_bf16(a0, w1, c01);
    c10 = wmma_bf16(a1, w0, c10);
    c11 = wmma_bf16(a1, w1, c11);
  }

  const v8f acc[4] = { c00, c01, c10, c11 };
  #pragma unroll
  for (int tt = 0; tt < 4; ++tt) {
    int tr = row0 + (tt >> 1) * 16;
    int tc = col0 + (tt & 1)  * 16;
    #pragma unroll
    for (int rI = 0; rI < 8; ++rI) {
      float v = acc[tt][rI];
      if (act == 1) v = tanhf(v);
      size_t idx = (size_t)(tr + rI + 8 * kh) * (size_t)N + (size_t)(tc + lc);
      if (Cf) Cf[idx] = v;
      if (Cb) Cb[idx] = f2bf(v);
    }
  }
}

// ------------- embedding gather (token -> padded bf16 rows) -------------
__global__ void embed_gather(const int* __restrict__ x, const float* __restrict__ tab,
                             bf16_t* __restrict__ emb)
{
  const int TW = 20, NW = 3200, E = 300, EP = 320, V = 51917, S = 20, B = 32, L = 2000;
  int m = blockIdx.x * blockDim.x + threadIdx.x;
  if (m >= TW * NW) return;
  int t = m / NW, n = m - t * NW;
  int i = n / B, b = n - i * B;
  int tok = x[b * L + i * S + t];
  tok = tok < 0 ? 0 : (tok > V - 1 ? V - 1 : tok);
  const float* row = tab + (size_t)tok * E;
  bf16_t* dst = emb + (size_t)m * EP;
  for (int e = 0; e < E; ++e) dst[e] = f2bf(row[e]);
  for (int e = E; e < EP; ++e) dst[e] = f2bf(0.0f);
}

// ------------- weight conversion -------------
__global__ void conv_pad_bf16(const float* __restrict__ src, bf16_t* __restrict__ dst,
                              int rows, int K, int Kp)
{
  int idx = blockIdx.x * blockDim.x + threadIdx.x;
  if (idx >= rows * Kp) return;
  int r = idx / Kp, k = idx - r * Kp;
  dst[idx] = (k < K) ? f2bf(src[(size_t)r * K + k]) : f2bf(0.0f);
}
__global__ void conv_trans_bf16(const float* __restrict__ src, bf16_t* __restrict__ dst,
                                int R, int C)  // src[R,C] -> dst[C,R]
{
  int idx = blockIdx.x * blockDim.x + threadIdx.x;
  if (idx >= R * C) return;
  int r = idx / C, c = idx - r * C;
  dst[(size_t)c * R + r] = f2bf(src[idx]);
}
__global__ void zero_u32(unsigned* __restrict__ p, long n)
{
  long i = (long)blockIdx.x * blockDim.x + threadIdx.x;
  if (i < n) p[i] = 0u;
}

// ------------- word-level GRU gate fusion (both directions) -------------
__global__ void gru_gate_word(const float* __restrict__ xp_f, const float* __restrict__ xp_b,
                              const float* __restrict__ hp_f, const float* __restrict__ hp_b,
                              float* __restrict__ h_f, float* __restrict__ h_b,
                              bf16_t* __restrict__ hb_f, bf16_t* __restrict__ hb_b,
                              bf16_t* __restrict__ out_w, int t)
{
  const int NW = 3200, H = 256, G = 768, T = 20;
  int idx = blockIdx.x * blockDim.x + threadIdx.x;
  if (idx >= 2 * NW * H) return;
  int dir = idx >= NW * H;
  int rem = idx - dir * NW * H;
  int n = rem / H, j = rem - n * H;
  const float* xp = dir ? xp_b : xp_f;
  const float* hp = dir ? hp_b : hp_f;
  float*  hf = dir ? h_b  : h_f;
  bf16_t* hb = dir ? hb_b : hb_f;
  int tx = dir ? (T - 1 - t) : t;
  size_t xbase = ((size_t)tx * NW + n) * G;
  float xr = xp[xbase + j], xz = xp[xbase + H + j], xn = xp[xbase + 2 * H + j];
  size_t hbase = (size_t)n * G;
  float hr = hp[hbase + j], hz = hp[hbase + H + j], hn = hp[hbase + 2 * H + j];
  float h = hf[(size_t)n * H + j];
  float rg = sigmoidf_(xr + hr);
  float zg = sigmoidf_(xz + hz);
  float ng = tanhf(xn + rg * hn);
  float hnew = (1.0f - zg) * ng + zg * h;
  hf[(size_t)n * H + j] = hnew;
  hb[(size_t)n * H + j] = f2bf(hnew);
  out_w[((size_t)tx * NW + n) * 512 + dir * H + j] = f2bf(hnew);
}

// ------------- persistent sentence-level BiGRU scan (1 workgroup) -------------
// hp tiles computed via WMMA into dynamic LDS (2 * 32*768 f32 = 192KB), gates fused.
__global__ void __launch_bounds__(256)
sent_gru_scan(const float* __restrict__ xs_f, const float* __restrict__ xs_b,
              const bf16_t* __restrict__ whh_f, const bf16_t* __restrict__ whh_b,
              const float* __restrict__ bhh_f, const float* __restrict__ bhh_b,
              float* __restrict__ h_f, float* __restrict__ h_b,
              bf16_t* __restrict__ hb_f, bf16_t* __restrict__ hb_b,
              bf16_t* __restrict__ out_s)
{
  const int Bn = 32, H = 256, G = 768, R = 100;
  extern __shared__ float hp_lds[];                 // [2][Bn*G]
  const int lane = threadIdx.x & 31;
  const int wave = threadIdx.x >> 5;
  const int kh = lane >> 4, lc = lane & 15;

  for (int step = 0; step < R; ++step) {
    // GEMM phase: 2 dirs * 2 Mtiles * 48 Ntiles = 192 WMMA tiles over 8 waves
    for (int tile = wave; tile < 192; tile += 8) {
      int dir = tile >= 96;
      int rem = tile - dir * 96;
      int mt = rem & 1, nt = rem >> 1;
      const bf16_t* Ah = dir ? hb_b : hb_f;
      const bf16_t* Wt = dir ? whh_b : whh_f;
      const float* bias = dir ? bhh_b : bhh_f;
      int row0 = mt * 16, col0 = nt * 16;
      float bv = bias[col0 + lc];
      v8f c;
      #pragma unroll
      for (int i = 0; i < 8; ++i) c[i] = bv;
      for (int k0 = 0; k0 < H; k0 += 32) {
        v16bf a = load_frag(Ah, H, row0, k0, lane);
        v16bf w = load_frag(Wt, H, col0, k0, lane);
        c = wmma_bf16(a, w, c);
      }
      float* dst = hp_lds + dir * (Bn * G);
      #pragma unroll
      for (int rI = 0; rI < 8; ++rI)
        dst[(row0 + rI + 8 * kh) * G + col0 + lc] = c[rI];
    }
    __syncthreads();
    // gate phase: 2*32*256 = 16384 elements
    for (int e = threadIdx.x; e < 2 * Bn * H; e += 256) {
      int dir = e >= Bn * H;
      int rem = e - dir * Bn * H;
      int n = rem / H, j = rem - n * H;
      const float* xs = dir ? xs_b : xs_f;
      float*  hf = dir ? h_b  : h_f;
      bf16_t* hb = dir ? hb_b : hb_f;
      int tx = dir ? (R - 1 - step) : step;
      size_t xbase = ((size_t)tx * Bn + n) * G;
      const float* hp = hp_lds + dir * (Bn * G) + n * G;
      float xr = xs[xbase + j], xz = xs[xbase + H + j], xn = xs[xbase + 2 * H + j];
      float hr = hp[j], hz = hp[H + j], hn = hp[2 * H + j];
      float h = hf[n * H + j];
      float rg = sigmoidf_(xr + hr);
      float zg = sigmoidf_(xz + hz);
      float ng = tanhf(xn + rg * hn);
      float hnew = (1.0f - zg) * ng + zg * h;
      hf[n * H + j] = hnew;
      hb[n * H + j] = f2bf(hnew);
      out_s[((size_t)tx * Bn + n) * 512 + dir * H + j] = f2bf(hnew);
    }
    __syncthreads();
  }
}

// ------------- attention pieces -------------
__global__ void rowdot_bf16(const bf16_t* __restrict__ Am, const float* __restrict__ v,
                            float* __restrict__ out, int M, int K)
{
  int m = blockIdx.x * blockDim.x + threadIdx.x;
  if (m >= M) return;
  const bf16_t* row = Am + (size_t)m * K;
  float s = 0.0f;
  for (int k = 0; k < K; ++k) s += bf2f(row[k]) * v[k];
  out[m] = s;
}
__global__ void softmax_axis0(float* __restrict__ sc, int T, int Ncol)
{
  int n = blockIdx.x * blockDim.x + threadIdx.x;
  if (n >= Ncol) return;
  float mx = -3.4e38f;
  for (int t = 0; t < T; ++t) { float v = sc[(size_t)t * Ncol + n]; mx = v > mx ? v : mx; }
  float sum = 0.0f;
  for (int t = 0; t < T; ++t) {
    float e = expf(sc[(size_t)t * Ncol + n] - mx);
    sc[(size_t)t * Ncol + n] = e; sum += e;
  }
  float inv = 1.0f / sum;
  for (int t = 0; t < T; ++t) sc[(size_t)t * Ncol + n] *= inv;
}
__global__ void attn_wsum(const float* __restrict__ attn, const bf16_t* __restrict__ seq,
                          float* __restrict__ vec, bf16_t* __restrict__ vecb,
                          int T, int Ncol, int Hd)
{
  int idx = blockIdx.x * blockDim.x + threadIdx.x;
  if (idx >= Ncol * Hd) return;
  int n = idx / Hd, h = idx - n * Hd;
  float s = 0.0f;
  for (int t = 0; t < T; ++t)
    s += attn[(size_t)t * Ncol + n] * bf2f(seq[((size_t)t * Ncol + n) * Hd + h]);
  vec[idx] = s;
  if (vecb) vecb[idx] = f2bf(s);
}
__global__ void final_linear(const float* __restrict__ doc, const float* __restrict__ Wf,
                             const float* __restrict__ bfv, float* __restrict__ out)
{
  const int Bn = 32, C = 50, K = 512;
  int idx = blockIdx.x * blockDim.x + threadIdx.x;
  if (idx >= Bn * C) return;
  int b = idx / C, c = idx - b * C;
  float s = bfv[c];
  const float* d = doc + (size_t)b * K;
  const float* w = Wf  + (size_t)c * K;
  for (int k = 0; k < K; ++k) s += d[k] * w[k];
  out[idx] = s;
}

// ------------- orchestration -------------
static inline dim3 grid1(long n, int tpb = 256) { return dim3((unsigned)((n + tpb - 1) / tpb)); }

extern "C" void kernel_launch(void* const* d_in, const int* in_sizes, int n_in,
                              void* d_out, int out_size, void* d_ws, size_t ws_size,
                              hipStream_t stream)
{
  (void)in_sizes; (void)n_in; (void)out_size; (void)ws_size;
  const int*   x          = (const int*)  d_in[0];
  const float* embed_tab  = (const float*)d_in[2];
  const float* wg_ih_f    = (const float*)d_in[3];
  const float* wg_hh_f    = (const float*)d_in[4];
  const float* wg_bih_f   = (const float*)d_in[5];
  const float* wg_bhh_f   = (const float*)d_in[6];
  const float* wg_ih_b    = (const float*)d_in[7];
  const float* wg_hh_b    = (const float*)d_in[8];
  const float* wg_bih_b   = (const float*)d_in[9];
  const float* wg_bhh_b   = (const float*)d_in[10];
  const float* W_word     = (const float*)d_in[11];
  const float* b_word     = (const float*)d_in[12];
  const float* proj_word  = (const float*)d_in[13];
  const float* sg_ih_f    = (const float*)d_in[14];
  const float* sg_hh_f    = (const float*)d_in[15];
  const float* sg_bih_f   = (const float*)d_in[16];
  const float* sg_bhh_f   = (const float*)d_in[17];
  const float* sg_ih_b    = (const float*)d_in[18];
  const float* sg_hh_b    = (const float*)d_in[19];
  const float* sg_bih_b   = (const float*)d_in[20];
  const float* sg_bhh_b   = (const float*)d_in[21];
  const float* W_sent     = (const float*)d_in[22];
  const float* b_sent     = (const float*)d_in[23];
  const float* proj_sent  = (const float*)d_in[24];
  const float* W_final    = (const float*)d_in[25];
  const float* b_final    = (const float*)d_in[26];
  float* out = (float*)d_out;

  const int TW = 20, NW = 3200, EP = 320, G = 768, H = 256, H2 = 512, R = 100, Bn = 32;
  const long MW = (long)TW * NW;   // 64000

  // bump allocator over d_ws (256B aligned)
  char* base = (char*)d_ws;
  size_t off = 0;
  auto alloc = [&](size_t bytes) -> void* {
    void* p = base + off;
    off = (off + bytes + 255) & ~(size_t)255;
    return p;
  };
  bf16_t* wgihF = (bf16_t*)alloc((size_t)G * EP * 2);
  bf16_t* wgihB = (bf16_t*)alloc((size_t)G * EP * 2);
  bf16_t* wghhF = (bf16_t*)alloc((size_t)G * H  * 2);
  bf16_t* wghhB = (bf16_t*)alloc((size_t)G * H  * 2);
  bf16_t* sgihF = (bf16_t*)alloc((size_t)G * H2 * 2);
  bf16_t* sgihB = (bf16_t*)alloc((size_t)G * H2 * 2);
  bf16_t* sghhF = (bf16_t*)alloc((size_t)G * H  * 2);
  bf16_t* sghhB = (bf16_t*)alloc((size_t)G * H  * 2);
  bf16_t* WwT   = (bf16_t*)alloc((size_t)H2 * H2 * 2);
  bf16_t* WsT   = (bf16_t*)alloc((size_t)H2 * H2 * 2);
  bf16_t* emb   = (bf16_t*)alloc((size_t)MW * EP * 2);
  float*  xpF   = (float*) alloc((size_t)MW * G * 4);
  float*  xpB   = (float*) alloc((size_t)MW * G * 4);
  float*  hpF   = (float*) alloc((size_t)NW * G * 4);
  float*  hpB   = (float*) alloc((size_t)NW * G * 4);
  float*  hwF   = (float*) alloc((size_t)NW * H * 4);
  float*  hwB   = (float*) alloc((size_t)NW * H * 4);
  bf16_t* hwbF  = (bf16_t*)alloc((size_t)NW * H * 2);
  bf16_t* hwbB  = (bf16_t*)alloc((size_t)NW * H * 2);
  bf16_t* outW  = (bf16_t*)alloc((size_t)MW * H2 * 2);
  bf16_t* sqW   = (bf16_t*)alloc((size_t)MW * H2 * 2);
  float*  scW   = (float*) alloc((size_t)MW * 4);
  float*  sentV = (float*) alloc((size_t)NW * H2 * 4);
  bf16_t* sentVb= (bf16_t*)alloc((size_t)NW * H2 * 2);
  float*  xsF   = (float*) alloc((size_t)NW * G * 4);
  float*  xsB   = (float*) alloc((size_t)NW * G * 4);
  float*  hsF   = (float*) alloc((size_t)Bn * H * 4);
  float*  hsB   = (float*) alloc((size_t)Bn * H * 4);
  bf16_t* hsbF  = (bf16_t*)alloc((size_t)Bn * H * 2);
  bf16_t* hsbB  = (bf16_t*)alloc((size_t)Bn * H * 2);
  bf16_t* outS  = (bf16_t*)alloc((size_t)NW * H2 * 2);
  bf16_t* sqS   = (bf16_t*)alloc((size_t)NW * H2 * 2);
  float*  scS   = (float*) alloc((size_t)NW * 4);
  float*  docV  = (float*) alloc((size_t)Bn * H2 * 4);

  // per-GEMM dynamic LDS: 128 columns x K bf16
  const size_t ldsEP = (size_t)128 * EP * 2;   // 80 KB
  const size_t ldsH  = (size_t)128 * H  * 2;   // 64 KB
  const size_t ldsH2 = (size_t)128 * H2 * 2;   // 128 KB

  // 1) weight conversions (BT layout = [N,K], K contiguous; E padded 300->320)
  conv_pad_bf16<<<grid1((long)G * EP), 256, 0, stream>>>(wg_ih_f, wgihF, G, 300, EP);
  conv_pad_bf16<<<grid1((long)G * EP), 256, 0, stream>>>(wg_ih_b, wgihB, G, 300, EP);
  conv_pad_bf16<<<grid1((long)G * H),  256, 0, stream>>>(wg_hh_f, wghhF, G, H, H);
  conv_pad_bf16<<<grid1((long)G * H),  256, 0, stream>>>(wg_hh_b, wghhB, G, H, H);
  conv_pad_bf16<<<grid1((long)G * H2), 256, 0, stream>>>(sg_ih_f, sgihF, G, H2, H2);
  conv_pad_bf16<<<grid1((long)G * H2), 256, 0, stream>>>(sg_ih_b, sgihB, G, H2, H2);
  conv_pad_bf16<<<grid1((long)G * H),  256, 0, stream>>>(sg_hh_f, sghhF, G, H, H);
  conv_pad_bf16<<<grid1((long)G * H),  256, 0, stream>>>(sg_hh_b, sghhB, G, H, H);
  conv_trans_bf16<<<grid1((long)H2 * H2), 256, 0, stream>>>(W_word, WwT, H2, H2);
  conv_trans_bf16<<<grid1((long)H2 * H2), 256, 0, stream>>>(W_sent, WsT, H2, H2);

  // 2) zero recurrent state
  zero_u32<<<grid1((long)NW * H), 256, 0, stream>>>((unsigned*)hwF,  (long)NW * H);
  zero_u32<<<grid1((long)NW * H), 256, 0, stream>>>((unsigned*)hwB,  (long)NW * H);
  zero_u32<<<grid1((long)NW * H / 2), 256, 0, stream>>>((unsigned*)hwbF, (long)NW * H / 2);
  zero_u32<<<grid1((long)NW * H / 2), 256, 0, stream>>>((unsigned*)hwbB, (long)NW * H / 2);
  zero_u32<<<grid1((long)Bn * H), 256, 0, stream>>>((unsigned*)hsF,  (long)Bn * H);
  zero_u32<<<grid1((long)Bn * H), 256, 0, stream>>>((unsigned*)hsB,  (long)Bn * H);
  zero_u32<<<grid1((long)Bn * H / 2), 256, 0, stream>>>((unsigned*)hsbF, (long)Bn * H / 2);
  zero_u32<<<grid1((long)Bn * H / 2), 256, 0, stream>>>((unsigned*)hsbB, (long)Bn * H / 2);

  // 3) embedding gather -> bf16 [20,3200,320]
  embed_gather<<<grid1(MW), 256, 0, stream>>>(x, embed_tab, emb);

  // 4) word input projections xp = emb @ Wih^T + bih    [64000 x 768]
  gemm_bf16_wmma<<<dim3(G / 128, (unsigned)(MW / 64)), 256, ldsEP, stream>>>(
      emb, EP, wgihF, EP, wg_bih_f, xpF, nullptr, G, EP, 0);
  gemm_bf16_wmma<<<dim3(G / 128, (unsigned)(MW / 64)), 256, ldsEP, stream>>>(
      emb, EP, wgihB, EP, wg_bih_b, xpB, nullptr, G, EP, 0);

  // 5) word-level BiGRU scan: per-step WMMA GEMM + fused gates
  for (int t = 0; t < TW; ++t) {
    gemm_bf16_wmma<<<dim3(G / 128, NW / 64), 256, ldsH, stream>>>(
        hwbF, H, wghhF, H, wg_bhh_f, hpF, nullptr, G, H, 0);
    gemm_bf16_wmma<<<dim3(G / 128, NW / 64), 256, ldsH, stream>>>(
        hwbB, H, wghhB, H, wg_bhh_b, hpB, nullptr, G, H, 0);
    gru_gate_word<<<grid1(2L * NW * H), 256, 0, stream>>>(
        xpF, xpB, hpF, hpB, hwF, hwB, hwbF, hwbB, outW, t);
  }

  // 6) word attention
  gemm_bf16_wmma<<<dim3(H2 / 128, (unsigned)(MW / 64)), 256, ldsH2, stream>>>(
      outW, H2, WwT, H2, b_word, nullptr, sqW, H2, H2, /*tanh*/1);
  rowdot_bf16<<<grid1(MW), 256, 0, stream>>>(sqW, proj_word, scW, (int)MW, H2);
  softmax_axis0<<<grid1(NW), 256, 0, stream>>>(scW, TW, NW);
  attn_wsum<<<grid1((long)NW * H2), 256, 0, stream>>>(scW, outW, sentV, sentVb, TW, NW, H2);

  // 7) sentence input projections xs = sent @ sgih^T + bih   [3200 x 768]
  gemm_bf16_wmma<<<dim3(G / 128, NW / 64), 256, ldsH2, stream>>>(
      sentVb, H2, sgihF, H2, sg_bih_f, xsF, nullptr, G, H2, 0);
  gemm_bf16_wmma<<<dim3(G / 128, NW / 64), 256, ldsH2, stream>>>(
      sentVb, H2, sgihB, H2, sg_bih_b, xsB, nullptr, G, H2, 0);

  // 8) persistent sentence BiGRU scan (1 workgroup, 100 steps, LDS hp)
  sent_gru_scan<<<1, 256, 2 * Bn * G * (int)sizeof(float), stream>>>(
      xsF, xsB, sghhF, sghhB, sg_bhh_f, sg_bhh_b, hsF, hsB, hsbF, hsbB, outS);

  // 9) sentence attention
  gemm_bf16_wmma<<<dim3(H2 / 128, NW / 64), 256, ldsH2, stream>>>(
      outS, H2, WsT, H2, b_sent, nullptr, sqS, H2, H2, /*tanh*/1);
  rowdot_bf16<<<grid1(NW), 256, 0, stream>>>(sqS, proj_sent, scS, NW, H2);
  softmax_axis0<<<grid1(Bn), 64, 0, stream>>>(scS, R, Bn);
  attn_wsum<<<grid1((long)Bn * H2), 256, 0, stream>>>(scS, outS, docV, nullptr, R, Bn, H2);

  // 10) classifier head
  final_linear<<<grid1((long)Bn * 50), 256, 0, stream>>>(docV, W_final, b_final, out);
}
